// DCNConv_70531952935435
// MI455X (gfx1250) — compile-verified, matrix operand order checked by reading, not run on hardware
//
#include <hip/hip_runtime.h>
#include <hip/hip_bf16.h>
#include <math.h>

// ---------------------------------------------------------------------------
// DCN block for MI455X (gfx1250, wave32, WMMA 16x16x32 bf16, fp32 accum).
// GEMMs with tap-major K (k = tap*Cin + ci). B tiles staged into LDS in
// fragment order (consumers do 2x ds_load_b128 per fragment). N tile = 64
// pixels; full M=256 per block: each wave owns 2 M-tiles x 4 N-subtiles
// -> 8 WMMAs per K-chunk per wave, B tile staged once for all 256 channels.
// LDS double-buffered: one barrier per chunk. Intermediate h is zero-padded
// (fp32: 84x84 halo2, bf16: 82x82 halo1) so offset-conv staging and the
// deform bilinear gather are branchless.
// ---------------------------------------------------------------------------

typedef __attribute__((ext_vector_type(16))) __bf16 v16bf;
typedef __attribute__((ext_vector_type(8)))  float v8f;

#define S32 84            // padded fp32 h row stride (80 + 2*2)
#define CH32 7056         // 84*84
#define S16 82            // padded bf16 h row stride (80 + 2*1)
#define CH16 6724         // 82*82

__device__ __forceinline__ unsigned short f2bf(float f) {
  unsigned int u = __builtin_bit_cast(unsigned int, f);
  unsigned int r = (u + 0x7FFFu + ((u >> 16) & 1u)) >> 16;   // RNE
  return (unsigned short)r;
}
__device__ __forceinline__ int klocal(int e, int half) {
  int i = e >> 1, r = e & 1;
  int kb = (i < 4) ? (i * 2) : (16 + (i - 4) * 2);
  return kb + r + half * 8;
}
// Inverse: (kl, n) -> ushort slot inside a 4x(32x16) fragment-ordered tile.
__device__ __forceinline__ int frag_slot(int kl, int n) {
  int half = (kl >> 3) & 1;
  int k2   = kl & ~8;
  int i    = (k2 < 8) ? (k2 >> 1) : (4 + ((k2 - 16) >> 1));
  int e    = 2 * i + (k2 & 1);
  int nsub = n >> 4, ncol = n & 15;
  return nsub * 512 + (half * 16 + ncol) * 16 + e;
}
__device__ __forceinline__ float silu(float v) { return v / (1.0f + __expf(-v)); }
__device__ __forceinline__ v8f wmma_bf16(v16bf a, v16bf b, v8f c) {
  return __builtin_amdgcn_wmma_f32_16x16x32_bf16(false, a, false, b, (short)0, c,
                                                 false, false);
}

// ---------------------------------------------------------------------------
// Pack OIHW fp32 weights into bf16 WMMA A fragments:
// dst[((mt*ktiles + kt)*32 + lane)*16 + e], k = tap*cin + ci. Rows >= coutValid
// are zero-padded (18 -> 32 pad for the offset conv).
// ---------------------------------------------------------------------------
__global__ void pack_w(const float* __restrict__ w, unsigned short* __restrict__ dst,
                       int coutValid, int cin, int mtiles, int ktiles) {
  int idx = blockIdx.x * blockDim.x + threadIdx.x;
  int total = mtiles * ktiles * 32 * 16;
  if (idx >= total) return;
  int e    = idx & 15;
  int lane = (idx >> 4) & 31;
  int rest = idx >> 9;
  int kt   = rest % ktiles;
  int mt   = rest / ktiles;
  int m    = mt * 16 + (lane & 15);
  int half = lane >> 4;
  int k    = kt * 32 + klocal(e, half);
  int t    = k / cin;
  int ci   = k - t * cin;
  float v = 0.0f;
  if (m < coutValid) v = w[(m * cin + ci) * 9 + t];
  dst[idx] = f2bf(v);
}

// ---------------------------------------------------------------------------
// GEMM1: conv1(3x3,s2,p1) 128->256 + BN1 + SiLU.
// Block: 256 thr = 8 waves, full M=256, N tile = 64 pixels.
// Wave w owns M-tiles {w, w+8}; 8 accumulators, 8 WMMAs per K-chunk.
// ---------------------------------------------------------------------------
__global__ __launch_bounds__(256) void conv1_gemm(
    const float* __restrict__ x, const unsigned short* __restrict__ apack,
    const float* __restrict__ g1, const float* __restrict__ b1,
    const float* __restrict__ m1, const float* __restrict__ v1,
    float* __restrict__ h32, unsigned short* __restrict__ h16) {
  __shared__ __align__(32) unsigned short bl[2][2048];
  const int tid  = threadIdx.x;
  const int lane = tid & 31, wave = tid >> 5;
  const int nbase = blockIdx.x * 64;
  const int ncol  = lane & 15, half = lane >> 4;

  int slot[8], klc[8], xb[8], py0[8], px0[8];
  #pragma unroll
  for (int i = 0; i < 8; ++i) {
    int idx = tid + i * 256;
    int n = idx & 63;
    klc[i]  = idx >> 6;
    slot[i] = frag_slot(klc[i], n);
    int p = nbase + n;
    int b = p / 6400, rem = p - b * 6400;
    int oy = rem / 80, ox = rem - oy * 80;
    xb[i]  = b * 3276800;              // b * 128 * 160 * 160
    py0[i] = oy * 2 - 1;
    px0[i] = ox * 2 - 1;
  }
  const unsigned short* ap0 = apack + (size_t)wave * 36 * 512 + lane * 16;
  const unsigned short* ap1 = ap0 + (size_t)8 * 36 * 512;

  v8f acc[2][4];
  #pragma unroll
  for (int j = 0; j < 2; ++j)
    #pragma unroll
    for (int ns = 0; ns < 4; ++ns) acc[j][ns] = (v8f){};

  for (int t = 0; t < 9; ++t) {
    const int kh = t / 3, kw = t % 3;
    int base[8]; float msk[8];
    #pragma unroll
    for (int i = 0; i < 8; ++i) {
      int iy = py0[i] + kh, ix = px0[i] + kw;
      bool ok = (unsigned)iy < 160u && (unsigned)ix < 160u;
      base[i] = ok ? (xb[i] + iy * 160 + ix) : xb[i];   // always-valid address
      msk[i]  = ok ? 1.0f : 0.0f;
    }
    #pragma unroll
    for (int c = 0; c < 4; ++c) {
      const int ks = t * 4 + c;
      unsigned short* dst = bl[c & 1];
      #pragma unroll
      for (int i = 0; i < 8; ++i) {
        float v = x[base[i] + ((c << 5) + klc[i]) * 25600] * msk[i];
        dst[slot[i]] = f2bf(v);
      }
      __syncthreads();
      v16bf a0 = *(const v16bf*)(ap0 + (size_t)ks * 512);
      v16bf a1 = *(const v16bf*)(ap1 + (size_t)ks * 512);
      const unsigned short* src = bl[c & 1] + lane * 16;
      #pragma unroll
      for (int ns = 0; ns < 4; ++ns) {
        v16bf bfr = *(const v16bf*)(src + ns * 512);
        acc[0][ns] = wmma_bf16(a0, bfr, acc[0][ns]);
        acc[1][ns] = wmma_bf16(a1, bfr, acc[1][ns]);
      }
    }
  }

  #pragma unroll
  for (int ns = 0; ns < 4; ++ns) {
    int p  = nbase + ns * 16 + ncol;
    int b  = p / 6400, rem = p - b * 6400;
    int oy = rem / 80, ox = rem - oy * 80;
    #pragma unroll
    for (int j = 0; j < 2; ++j) {
      #pragma unroll
      for (int r = 0; r < 8; ++r) {
        int co = j * 128 + wave * 16 + r + half * 8;
        float s  = g1[co] * rsqrtf(v1[co] + 1e-5f);
        float vv = (acc[j][ns][r] - m1[co]) * s + b1[co];
        vv = silu(vv);
        h32[(b * 256 + co) * CH32 + (oy + 2) * S32 + (ox + 2)] = vv;
        h16[(b * 256 + co) * CH16 + (oy + 1) * S16 + (ox + 1)] = f2bf(vv);
      }
    }
  }
}

// ---------------------------------------------------------------------------
// GEMM2: offset conv (3x3,s1,p1) 256->18 (M padded to 32) + bias.
// Reads the 1-px zero-padded bf16 h: staging is fully branchless.
// ---------------------------------------------------------------------------
__global__ __launch_bounds__(256) void off_gemm(
    const unsigned short* __restrict__ h16, const unsigned short* __restrict__ apack,
    const float* __restrict__ obias, float* __restrict__ offs) {
  __shared__ __align__(32) unsigned short bl[2][2048];
  const int tid  = threadIdx.x;
  const int lane = tid & 31, wave = tid >> 5;
  const int mt   = wave >> 2, nsub = wave & 3;
  const int nbase = blockIdx.x * 64;
  const int ncol  = lane & 15, half = lane >> 4;

  int slot[8], klc[8], pb[8], oyv[8], oxv[8];
  #pragma unroll
  for (int i = 0; i < 8; ++i) {
    int idx = tid + i * 256;
    int n = idx & 63;
    klc[i]  = idx >> 6;
    slot[i] = frag_slot(klc[i], n);
    int p = nbase + n;
    int b = p / 6400, rem = p - b * 6400;
    oyv[i] = rem / 80; oxv[i] = rem - oyv[i] * 80;
    pb[i]  = b * 256 * CH16;
  }
  const unsigned short* ap = apack + (size_t)mt * 72 * 512 + lane * 16;

  v8f acc = {};
  for (int t = 0; t < 9; ++t) {
    const int kh = t / 3, kw = t % 3;
    int base[8];
    #pragma unroll
    for (int i = 0; i < 8; ++i)       // (oy-1+kh)+1 in [0,81]: always in-bounds
      base[i] = pb[i] + (oyv[i] + kh) * S16 + (oxv[i] + kw);
    #pragma unroll
    for (int c = 0; c < 8; ++c) {
      const int ks = t * 8 + c;
      unsigned short* dst = bl[c & 1];
      #pragma unroll
      for (int i = 0; i < 8; ++i)
        dst[slot[i]] = h16[base[i] + ((c << 5) + klc[i]) * CH16];
      __syncthreads();
      v16bf a = *(const v16bf*)(ap + (size_t)ks * 512);
      v16bf bfr = *(const v16bf*)(bl[c & 1] + nsub * 512 + lane * 16);
      acc = wmma_bf16(a, bfr, acc);
    }
  }

  int p  = nbase + nsub * 16 + ncol;
  int b  = p / 6400, rem = p - b * 6400;
  int oy = rem / 80, ox = rem - oy * 80;
  #pragma unroll
  for (int r = 0; r < 8; ++r) {
    int co = mt * 16 + r + half * 8;
    if (co < 18)
      offs[((b * 18 + co) * 80 + oy) * 80 + ox] = acc[r] + obias[co];
  }
}

// ---------------------------------------------------------------------------
// GEMM3: deformable 3x3 conv 256->256 + BN2 + SiLU, full M=256 per block.
// Geometry per (tap, pixel): one base address into the 2-px zero-padded fp32 h
// (coords clamped to [-2,80] -> clamped/OOB corners read the zero halo) plus 4
// precomputed bilinear weights. Inner gather: 4 loads at immediate offsets
// {0,1,S32,S32+1} + 4 FMAs, branchless; gathered once for all 256 channels.
// ---------------------------------------------------------------------------
__global__ __launch_bounds__(256) void dcn_gemm(
    const float* __restrict__ h32, const float* __restrict__ offs,
    const unsigned short* __restrict__ apack,
    const float* __restrict__ g2, const float* __restrict__ b2,
    const float* __restrict__ m2, const float* __restrict__ v2,
    float* __restrict__ out) {
  __shared__ __align__(32) unsigned short bl[2][2048];
  __shared__ int   ga[576];
  __shared__ float gw00[576], gw01[576], gw10[576], gw11[576];
  const int tid  = threadIdx.x;
  const int lane = tid & 31, wave = tid >> 5;
  const int nbase = blockIdx.x * 64;
  const int ncol  = lane & 15, half = lane >> 4;

  // geometry: entry gi = t*64 + n
  for (int gi = tid; gi < 576; gi += 256) {
    int n = gi & 63, t = gi >> 6;
    int p  = nbase + n;
    int b  = p / 6400, rem = p - b * 6400;
    int oy = rem / 80, ox = rem - oy * 80;
    int kh = t / 3, kw = t % 3;
    float offY = offs[((b * 18 + 2 * t    ) * 80 + oy) * 80 + ox];
    float offX = offs[((b * 18 + 2 * t + 1) * 80 + oy) * 80 + ox];
    float py = (float)(oy + kh - 1) + offY;
    float px = (float)(ox + kw - 1) + offX;
    float fy = floorf(py), fx = floorf(px);
    float wy = py - fy,    wx = px - fx;
    int y0 = min(max((int)fy, -2), 80);     // clamped corners land in zero halo
    int x0 = min(max((int)fx, -2), 80);
    ga[gi]   = (y0 + 2) * S32 + (x0 + 2);
    gw00[gi] = (1.0f - wy) * (1.0f - wx);
    gw01[gi] = (1.0f - wy) * wx;
    gw10[gi] = wy * (1.0f - wx);
    gw11[gi] = wy * wx;
  }

  int slot[8], klc[8], nn[8], hb[8];
  #pragma unroll
  for (int i = 0; i < 8; ++i) {
    int idx = tid + i * 256;
    nn[i]   = idx & 63;
    klc[i]  = idx >> 6;
    slot[i] = frag_slot(klc[i], nn[i]);
    int p = nbase + nn[i];
    int b = p / 6400;
    hb[i] = b * 256 * CH32;
  }
  const unsigned short* ap0 = apack + (size_t)wave * 72 * 512 + lane * 16;
  const unsigned short* ap1 = ap0 + (size_t)8 * 72 * 512;
  __syncthreads();

  v8f acc[2][4];
  #pragma unroll
  for (int j = 0; j < 2; ++j)
    #pragma unroll
    for (int ns = 0; ns < 4; ++ns) acc[j][ns] = (v8f){};

  for (int t = 0; t < 9; ++t) {
    int   a0[8];
    float w00[8], w01[8], w10[8], w11[8];
    #pragma unroll
    for (int i = 0; i < 8; ++i) {
      int gi = (t << 6) + nn[i];
      a0[i] = ga[gi];
      w00[i] = gw00[gi]; w01[i] = gw01[gi];
      w10[i] = gw10[gi]; w11[i] = gw11[gi];
    }
    #pragma unroll
    for (int c = 0; c < 8; ++c) {
      const int ks = t * 8 + c;
      unsigned short* dst = bl[c & 1];
      #pragma unroll
      for (int i = 0; i < 8; ++i) {
        const float* hc = h32 + hb[i] + ((c << 5) + klc[i]) * CH32 + a0[i];
        float s = w00[i] * hc[0] + w01[i] * hc[1]
                + w10[i] * hc[S32] + w11[i] * hc[S32 + 1];
        dst[slot[i]] = f2bf(s);
      }
      __syncthreads();
      v16bf a0f = *(const v16bf*)(ap0 + (size_t)ks * 512);
      v16bf a1f = *(const v16bf*)(ap1 + (size_t)ks * 512);
      const unsigned short* src = bl[c & 1] + lane * 16;
      #pragma unroll
      for (int ns = 0; ns < 4; ++ns) {
        v16bf bfr = *(const v16bf*)(src + ns * 512);
        acc[0][ns] = wmma_bf16(a0f, bfr, acc[0][ns]);
        acc[1][ns] = wmma_bf16(a1f, bfr, acc[1][ns]);
      }
    }
  }

  #pragma unroll
  for (int ns = 0; ns < 4; ++ns) {
    int p  = nbase + ns * 16 + ncol;
    int b  = p / 6400, rem = p - b * 6400;
    int oy = rem / 80, ox = rem - oy * 80;
    #pragma unroll
    for (int j = 0; j < 2; ++j) {
      #pragma unroll
      for (int r = 0; r < 8; ++r) {
        int co = j * 128 + wave * 16 + r + half * 8;
        float s  = g2[co] * rsqrtf(v2[co] + 1e-5f);
        float vv = (acc[j][ns][r] - m2[co]) * s + b2[co];
        out[((b * 256 + co) * 80 + oy) * 80 + ox] = silu(vv);
      }
    }
  }
}

// ---------------------------------------------------------------------------
extern "C" void kernel_launch(void* const* d_in, const int* in_sizes, int n_in,
                              void* d_out, int out_size, void* d_ws, size_t ws_size,
                              hipStream_t stream) {
  const float* x       = (const float*)d_in[0];
  const float* conv1_w = (const float*)d_in[1];
  const float* bn1_g   = (const float*)d_in[2];
  const float* bn1_b   = (const float*)d_in[3];
  const float* bn1_m   = (const float*)d_in[4];
  const float* bn1_v   = (const float*)d_in[5];
  const float* off_w   = (const float*)d_in[6];
  const float* off_b   = (const float*)d_in[7];
  const float* dconv_w = (const float*)d_in[8];
  const float* bn2_g   = (const float*)d_in[9];
  const float* bn2_b   = (const float*)d_in[10];
  const float* bn2_m   = (const float*)d_in[11];
  const float* bn2_v   = (const float*)d_in[12];
  float* out = (float*)d_out;

  char* ws = (char*)d_ws;
  size_t o = 0;
  auto carve = [&](size_t bytes) {
    void* p = ws + o;
    o = (o + bytes + 255) & ~(size_t)255;
    return p;
  };
  unsigned short* a1   = (unsigned short*)carve((size_t)16 * 36 * 512 * 2);
  unsigned short* a2   = (unsigned short*)carve((size_t)2  * 72 * 512 * 2);
  unsigned short* a3   = (unsigned short*)carve((size_t)16 * 72 * 512 * 2);
  size_t h32_bytes = (size_t)8 * 256 * CH32 * 4;
  size_t h16_bytes = (size_t)8 * 256 * CH16 * 2;
  float*          hp32 = (float*)         carve(h32_bytes);
  unsigned short* hp16 = (unsigned short*)carve(h16_bytes);
  float*          ofs  = (float*)         carve((size_t)8 * 18 * 6400 * 4);

  // zero the padded h buffers (halos must be 0; interiors rewritten below)
  hipMemsetAsync(hp32, 0, h32_bytes, stream);
  hipMemsetAsync(hp16, 0, h16_bytes, stream);

  {
    int n1 = 16 * 36 * 512;
    pack_w<<<(n1 + 255) / 256, 256, 0, stream>>>(conv1_w, a1, 256, 128, 16, 36);
    int n2 = 2 * 72 * 512;
    pack_w<<<(n2 + 255) / 256, 256, 0, stream>>>(off_w,   a2, 18,  256, 2,  72);
    int n3 = 16 * 72 * 512;
    pack_w<<<(n3 + 255) / 256, 256, 0, stream>>>(dconv_w, a3, 256, 256, 16, 72);
  }

  const int nblk = (8 * 80 * 80) / 64;   // 800
  conv1_gemm<<<nblk, 256, 0, stream>>>(x, a1, bn1_g, bn1_b, bn1_m, bn1_v, hp32, hp16);
  off_gemm<<<nblk, 256, 0, stream>>>(hp16, a2, off_b, ofs);
  dcn_gemm<<<nblk, 256, 0, stream>>>(hp32, ofs, a3, bn2_g, bn2_b, bn2_m, bn2_v, out);
}
